// DecoderLayer_5351529251311
// MI455X (gfx1250) — matrix operand order repaired and checked by reading, not compiled, not run on hardware
//
#include <hip/hip_runtime.h>
#include <math.h>

// ---------------------------------------------------------------------------
// Types for CDNA5 WMMA (wave32): v_wmma_f32_16x16x32_bf16
// ---------------------------------------------------------------------------
typedef __attribute__((ext_vector_type(16))) __bf16 bf16x16;
typedef __attribute__((ext_vector_type(8)))  __bf16 bf16x8;
typedef __attribute__((ext_vector_type(8)))  float  f32x8;

// ---------------------------------------------------------------------------
// gfx1250 async global->LDS copy (ASYNCcnt path) via inline asm.
// The clang builtin exists but its pointer params are cuda_device/cuda_shared
// qualified (unspellable from HIP source), so we emit the instruction
// directly. VDST = per-lane LDS byte offset, VADDR = 64-bit global address.
// ---------------------------------------------------------------------------
__device__ __forceinline__ void async_copy_b128(const __bf16* __restrict__ g,
                                                __bf16* __restrict__ l)
{
#if defined(__AMDGCN__)
    unsigned loff = (unsigned)(unsigned long long)
        (__attribute__((address_space(3))) const void*)l;   // LDS byte offset
    asm volatile("global_load_async_to_lds_b128 %0, %1, off"
                 :
                 : "v"(loff), "v"(g)
                 : "memory");
#else
    *(bf16x8*)l = *(const bf16x8*)g;
#endif
}

__device__ __forceinline__ void wait_async_zero()
{
#if defined(__AMDGCN__)
    asm volatile("s_wait_asynccnt 0x0" ::: "memory");
#endif
}

// Problem constants (fixed by the reference)
constexpr int Bc     = 2;
constexpr int Sc     = 2048;
constexpr int Dc     = 1024;
constexpr int Hc     = 16;
constexpr int DEPTHc = 64;
constexpr int DFFc   = 4096;
constexpr int BSc    = Bc * Sc;                                  // 4096 tokens
constexpr long long X2_ELEMS = (long long)BSc * Dc;              // 4194304

// GEMM tiling
constexpr int BM = 128;
constexpr int BN = 128;
constexpr int BKt = 32;
constexpr int LDSROW = 48;   // 32 + 16 pad -> 96B row stride, keeps b128 LDS ops aligned

// Epilogue modes
// 0: bias + bf16 out, head-split [B,S,D] -> [B*H, S, 64]          (Q,K)
// 1: scale + causal mask, f32 out to attn[z]                      (scores)
// 2: bf16 out, head-merge [B*H,S,64] -> [B,S,D]                   (ctx)
// 3: bias + residual(f32) -> f32 out [M,N]
// 4: bias + exact GELU -> bf16 out [M,N]
// 5: bias + bf16 out, head-split TRANSPOSED -> [B*H, 64, S]       (V)

template<int AMODE /*0=bf16 A, 1=f32 A*/, int EPI>
__global__ __launch_bounds__(256)
void gemm_wmma(const void* __restrict__ Agv,
               const __bf16* __restrict__ Bg,     // always N-major, K contiguous (ldb = row stride)
               void* __restrict__ Cgv,
               const float* __restrict__ bias,
               const float* __restrict__ resid,
               int M, int N, int K, int lda, int ldb,
               long long strideA, long long strideB, long long strideC,
               float scale)
{
    __shared__ __bf16 As[BM * LDSROW];
    __shared__ __bf16 Bs[BN * LDSROW];

    const int tid  = threadIdx.x;
    const int lane = tid & 31;
    const int wid  = tid >> 5;
    const int wr   = wid >> 2;   // 0..1
    const int wc   = wid & 3;    // 0..3
    const int z    = blockIdx.z;
    const int m0   = blockIdx.y * BM;
    const int n0   = blockIdx.x * BN;

    const __bf16* Ab = (AMODE == 0) ? ((const __bf16*)Agv + (long long)z * strideA) : nullptr;
    const float*  Af = (AMODE == 1) ? ((const float*)Agv  + (long long)z * strideA) : nullptr;
    const __bf16* Bz = Bg + (long long)z * strideB;

    f32x8 acc[4][2];
#pragma unroll
    for (int mt = 0; mt < 4; ++mt)
#pragma unroll
        for (int nt = 0; nt < 2; ++nt)
#pragma unroll
            for (int r = 0; r < 8; ++r) acc[mt][nt][r] = 0.0f;

    for (int k0 = 0; k0 < K; k0 += BKt) {
        __syncthreads();   // WAR: previous iteration's fragment reads done
        // ---- stage A tile: 128x32 bf16, 512 chunks of 16B ----
#pragma unroll
        for (int i = 0; i < 2; ++i) {
            int c  = tid + i * 256;       // 0..511
            int r  = c >> 2;              // 0..127
            int kc = (c & 3) << 3;        // 0,8,16,24
            __bf16* ldst = &As[r * LDSROW + kc];
            if (AMODE == 0) {
                const __bf16* p = Ab + (long long)(m0 + r) * lda + (k0 + kc);
                async_copy_b128(p, ldst);          // global -> LDS, ASYNCcnt
            } else {
                const float* p = Af + (long long)(m0 + r) * lda + (k0 + kc);
                float4 f0 = *(const float4*)p;
                float4 f1 = *(const float4*)(p + 4);
                bf16x8 v;
                v[0]=(__bf16)f0.x; v[1]=(__bf16)f0.y; v[2]=(__bf16)f0.z; v[3]=(__bf16)f0.w;
                v[4]=(__bf16)f1.x; v[5]=(__bf16)f1.y; v[6]=(__bf16)f1.z; v[7]=(__bf16)f1.w;
                *(bf16x8*)ldst = v;
            }
        }
        // ---- stage B tile (N-major, K contiguous in both global and LDS) ----
#pragma unroll
        for (int i = 0; i < 2; ++i) {
            int c  = tid + i * 256;       // 0..511
            int n  = c & 127;             // 0..127
            int kc = (c >> 7) << 3;       // 0,8,16,24
            int gn = n0 + n;
            __bf16* ldst = &Bs[n * LDSROW + kc];
            if (gn < N) {
                const __bf16* p = Bz + (long long)gn * ldb + (k0 + kc);
                async_copy_b128(p, ldst);          // global -> LDS, ASYNCcnt
            } else {
                bf16x8 zv;
#pragma unroll
                for (int j = 0; j < 8; ++j) zv[j] = (__bf16)0.0f;
                *(bf16x8*)ldst = zv;
            }
        }
        wait_async_zero();   // drain ASYNCcnt before the tile barrier
        __syncthreads();

        // ---- fragments per ISA 7.12.2 layouts, 8 WMMAs per K-step ----
        // A (16-bit 16x32): lanes 0-15 -> K {0..7,16..23}; lanes 16-31 -> K {8..15,24..31}
        const int akh = (lane < 16) ? 0 : 8;
        bf16x16 afr[4];
#pragma unroll
        for (int mt = 0; mt < 4; ++mt) {
            int r = wr * 64 + mt * 16 + (lane & 15);
            bf16x8 lo = *(const bf16x8*)&As[r * LDSROW + akh];
            bf16x8 hi = *(const bf16x8*)&As[r * LDSROW + akh + 16];
            afr[mt] = __builtin_shufflevector(lo, hi, 0,1,2,3,4,5,6,7,8,9,10,11,12,13,14,15);
        }
        // B (16-bit 32x16): lanes 0-15 -> K {0..7,16..23}; lanes 16-31 -> K {8..15,24..31}
        const int bkh = (lane < 16) ? 0 : 8;
        bf16x16 bfr[2];
#pragma unroll
        for (int nt = 0; nt < 2; ++nt) {
            int n = wc * 32 + nt * 16 + (lane & 15);
            bf16x8 lo = *(const bf16x8*)&Bs[n * LDSROW + bkh];
            bf16x8 hi = *(const bf16x8*)&Bs[n * LDSROW + bkh + 16];
            bfr[nt] = __builtin_shufflevector(lo, hi, 0,1,2,3,4,5,6,7,8,9,10,11,12,13,14,15);
        }
#pragma unroll
        for (int mt = 0; mt < 4; ++mt)
#pragma unroll
            for (int nt = 0; nt < 2; ++nt)
                acc[mt][nt] = __builtin_amdgcn_wmma_f32_16x16x32_bf16(
                    false, afr[mt], false, bfr[nt], (short)0, acc[mt][nt], false, false);
    }

    // ---- epilogue: C/D layout => VGPR r holds M = r (lanes 0-15) / 8+r (lanes 16-31)
    const int nb = n0 + wc * 32;
    const int mb = m0 + wr * 64 + ((lane < 16) ? 0 : 8);
    const int nl = lane & 15;
#pragma unroll
    for (int mt = 0; mt < 4; ++mt) {
#pragma unroll
        for (int nt = 0; nt < 2; ++nt) {
            const int gn = nb + nt * 16 + nl;
#pragma unroll
            for (int r = 0; r < 8; ++r) {
                const int gm = mb + mt * 16 + r;
                if (gm >= M || gn >= N) continue;
                float v = acc[mt][nt][r];
                if (EPI == 0) {          // Q/K: bias + head split -> bf16 [B*H, S, 64]
                    v += bias[gn];
                    int b = gm >> 11, s = gm & (Sc - 1);
                    int h = gn >> 6,  dd = gn & 63;
                    ((__bf16*)Cgv)[(((long long)(b * Hc + h)) * Sc + s) * DEPTHc + dd] = (__bf16)v;
                } else if (EPI == 1) {   // scores: scale + causal mask -> f32
                    v *= scale;
                    if (gn > gm) v = -1.0e9f;
                    ((float*)Cgv + (long long)z * strideC)[(long long)gm * Sc + gn] = v;
                } else if (EPI == 2) {   // ctx: head merge -> bf16 [B, S, D]
                    int b = z >> 4, h = z & 15;
                    ((__bf16*)Cgv)[((long long)(b * Sc + gm)) * Dc + h * DEPTHc + gn] = (__bf16)v;
                } else if (EPI == 3) {   // bias + residual -> f32
                    v += bias[gn] + resid[(long long)gm * N + gn];
                    ((float*)Cgv)[(long long)gm * N + gn] = v;
                } else if (EPI == 4) {   // bias + exact GELU -> bf16
                    v += bias[gn];
                    float g = 0.5f * v * (1.0f + erff(v * 0.70710678118654752f));
                    ((__bf16*)Cgv)[(long long)gm * N + gn] = (__bf16)g;
                } else if (EPI == 5) {   // V: bias + head split, transposed -> bf16 [B*H, 64, S]
                    v += bias[gn];
                    int b = gm >> 11, s = gm & (Sc - 1);
                    int h = gn >> 6,  dd = gn & 63;
                    ((__bf16*)Cgv)[(((long long)(b * Hc + h)) * DEPTHc + dd) * Sc + s] = (__bf16)v;
                }
            }
        }
    }
}

// ---------------------------------------------------------------------------
// f32 -> bf16 elementwise convert
// ---------------------------------------------------------------------------
__global__ __launch_bounds__(256)
void cvt_f32_bf16(const float* __restrict__ in, __bf16* __restrict__ out, long long n)
{
    long long i = (long long)blockIdx.x * blockDim.x + threadIdx.x;
    long long stride = (long long)gridDim.x * blockDim.x;
    for (; i < n; i += stride) out[i] = (__bf16)in[i];
}

// ---------------------------------------------------------------------------
// Tiled f32[K,N] -> bf16[N,K] transpose-convert (weights; K,N multiples of 32)
// ---------------------------------------------------------------------------
__global__ __launch_bounds__(256)
void transpose_cvt(const float* __restrict__ in, __bf16* __restrict__ out, int K, int N)
{
    __shared__ float tile[32][33];
    const int nb = blockIdx.x * 32;
    const int kb = blockIdx.y * 32;
    const int tx = threadIdx.x & 31;
    const int ty = threadIdx.x >> 5;     // 0..7
#pragma unroll
    for (int i = 0; i < 32; i += 8)
        tile[ty + i][tx] = in[(long long)(kb + ty + i) * N + (nb + tx)];
    __syncthreads();
#pragma unroll
    for (int i = 0; i < 32; i += 8)
        out[(long long)(nb + ty + i) * K + (kb + tx)] = (__bf16)tile[tx][ty + i];
}

// ---------------------------------------------------------------------------
// Row softmax over 2048-wide rows (in place, f32)
// ---------------------------------------------------------------------------
__global__ __launch_bounds__(256)
void softmax_rows(float* __restrict__ attn)
{
    __shared__ float red[256];
    float* p = attn + (long long)blockIdx.x * Sc;
    const int tid = threadIdx.x;

    float vals[8];
    float mx = -3.4e38f;
#pragma unroll
    for (int i = 0; i < 8; ++i) {
        vals[i] = p[tid + i * 256];
        mx = fmaxf(mx, vals[i]);
    }
    red[tid] = mx; __syncthreads();
    for (int s = 128; s > 0; s >>= 1) {
        if (tid < s) red[tid] = fmaxf(red[tid], red[tid + s]);
        __syncthreads();
    }
    mx = red[0]; __syncthreads();

    float sum = 0.0f;
#pragma unroll
    for (int i = 0; i < 8; ++i) {
        vals[i] = expf(vals[i] - mx);
        sum += vals[i];
    }
    red[tid] = sum; __syncthreads();
    for (int s = 128; s > 0; s >>= 1) {
        if (tid < s) red[tid] += red[tid + s];
        __syncthreads();
    }
    const float inv = 1.0f / red[0];
#pragma unroll
    for (int i = 0; i < 8; ++i) p[tid + i * 256] = vals[i] * inv;
}

// ---------------------------------------------------------------------------
// LayerNorm over 1024-wide rows; writes f32 and optionally bf16 copy
// ---------------------------------------------------------------------------
__global__ __launch_bounds__(256)
void layernorm_rows(const float* __restrict__ in,
                    const float* __restrict__ gamma,
                    const float* __restrict__ beta,
                    float* __restrict__ outf,
                    __bf16* __restrict__ outb)
{
    __shared__ float r1[256];
    __shared__ float r2[256];
    const long long row = blockIdx.x;
    const float* p = in + row * Dc;
    const int tid = threadIdx.x;

    float v[4];
    float s = 0.0f, s2 = 0.0f;
#pragma unroll
    for (int i = 0; i < 4; ++i) {
        float x = p[tid + i * 256];
        v[i] = x; s += x; s2 += x * x;
    }
    r1[tid] = s; r2[tid] = s2; __syncthreads();
    for (int w = 128; w > 0; w >>= 1) {
        if (tid < w) { r1[tid] += r1[tid + w]; r2[tid] += r2[tid + w]; }
        __syncthreads();
    }
    const float mu   = r1[0] * (1.0f / Dc);
    const float var  = r2[0] * (1.0f / Dc) - mu * mu;
    const float rstd = rsqrtf(var + 1e-5f);
#pragma unroll
    for (int i = 0; i < 4; ++i) {
        int col = tid + i * 256;
        float y = (v[i] - mu) * rstd * gamma[col] + beta[col];
        outf[row * Dc + col] = y;
        if (outb) outb[row * Dc + col] = (__bf16)y;
    }
}

// ---------------------------------------------------------------------------
// Host orchestration
// ---------------------------------------------------------------------------
extern "C" void kernel_launch(void* const* d_in, const int* in_sizes, int n_in,
                              void* d_out, int out_size, void* d_ws, size_t ws_size,
                              hipStream_t stream)
{
    (void)in_sizes; (void)n_in; (void)out_size; (void)ws_size;

    const float* x    = (const float*)d_in[0];
    // d_in[1] = mask (int32) — causal structure fused into the scores epilogue
    const float* Wq   = (const float*)d_in[2];
    const float* bq   = (const float*)d_in[3];
    const float* Wk   = (const float*)d_in[4];
    const float* bk   = (const float*)d_in[5];
    const float* Wv   = (const float*)d_in[6];
    const float* bv   = (const float*)d_in[7];
    const float* Wo   = (const float*)d_in[8];
    const float* bo   = (const float*)d_in[9];
    const float* W1   = (const float*)d_in[10];
    const float* b1   = (const float*)d_in[11];
    const float* W2   = (const float*)d_in[12];
    const float* b2   = (const float*)d_in[13];
    const float* ln1g = (const float*)d_in[14];
    const float* ln1b = (const float*)d_in[15];
    const float* ln2g = (const float*)d_in[16];
    const float* ln2b = (const float*)d_in[17];

    float* x2_out   = (float*)d_out;                 // [B,S,D]
    float* attn_out = (float*)d_out + X2_ELEMS;      // [B,H,S,S]

    // Workspace carve-out (256B aligned)
    char* wp = (char*)d_ws;
    auto carve = [&](size_t bytes) -> void* {
        void* p = (void*)wp;
        wp += (bytes + 255) & ~(size_t)255;
        return p;
    };
    __bf16* xb   = (__bf16*)carve((size_t)BSc * Dc * 2);
    __bf16* Wqt  = (__bf16*)carve((size_t)Dc * Dc * 2);     // [N=D, K=D] transposed bf16
    __bf16* Wkt  = (__bf16*)carve((size_t)Dc * Dc * 2);
    __bf16* Wvt  = (__bf16*)carve((size_t)Dc * Dc * 2);
    __bf16* Wot  = (__bf16*)carve((size_t)Dc * Dc * 2);
    __bf16* W1t  = (__bf16*)carve((size_t)Dc * DFFc * 2);   // [DFF, D]
    __bf16* W2t  = (__bf16*)carve((size_t)DFFc * Dc * 2);   // [D, DFF]
    __bf16* qb   = (__bf16*)carve((size_t)Bc * Hc * Sc * DEPTHc * 2);  // [B*H, S, 64]
    __bf16* kb   = (__bf16*)carve((size_t)Bc * Hc * Sc * DEPTHc * 2);  // [B*H, S, 64]
    __bf16* vt   = (__bf16*)carve((size_t)Bc * Hc * DEPTHc * Sc * 2);  // [B*H, 64, S]
    __bf16* ctxb = (__bf16*)carve((size_t)BSc * Dc * 2);
    float*  h1   = (float*)carve((size_t)BSc * Dc * 4);
    float*  x1f  = (float*)carve((size_t)BSc * Dc * 4);
    __bf16* x1b  = (__bf16*)carve((size_t)BSc * Dc * 2);
    __bf16* gb   = (__bf16*)carve((size_t)BSc * DFFc * 2);
    float*  h2   = (float*)carve((size_t)BSc * Dc * 4);

    const dim3 blk(256);

    // ---- activations -> bf16; weights -> transposed bf16 (N-major, K contiguous) ----
    cvt_f32_bf16<<<4096, blk, 0, stream>>>(x, xb, (long long)BSc * Dc);
    transpose_cvt<<<dim3(Dc / 32,  Dc / 32),  blk, 0, stream>>>(Wq, Wqt, Dc,  Dc);
    transpose_cvt<<<dim3(Dc / 32,  Dc / 32),  blk, 0, stream>>>(Wk, Wkt, Dc,  Dc);
    transpose_cvt<<<dim3(Dc / 32,  Dc / 32),  blk, 0, stream>>>(Wv, Wvt, Dc,  Dc);
    transpose_cvt<<<dim3(Dc / 32,  Dc / 32),  blk, 0, stream>>>(Wo, Wot, Dc,  Dc);
    transpose_cvt<<<dim3(DFFc / 32, Dc / 32), blk, 0, stream>>>(W1, W1t, Dc,  DFFc);
    transpose_cvt<<<dim3(Dc / 32, DFFc / 32), blk, 0, stream>>>(W2, W2t, DFFc, Dc);

    // ---- QKV projections: [4096,1024] @ W^T + bias ----
    {
        dim3 grid(Dc / BN, BSc / BM, 1);
        gemm_wmma<0,0><<<grid, blk, 0, stream>>>(xb, Wqt, qb, bq, nullptr,
            BSc, Dc, Dc, Dc, Dc, 0, 0, 0, 0.0f);
        gemm_wmma<0,0><<<grid, blk, 0, stream>>>(xb, Wkt, kb, bk, nullptr,
            BSc, Dc, Dc, Dc, Dc, 0, 0, 0, 0.0f);
        gemm_wmma<0,5><<<grid, blk, 0, stream>>>(xb, Wvt, vt, bv, nullptr,
            BSc, Dc, Dc, Dc, Dc, 0, 0, 0, 0.0f);
    }

    // ---- scores = Q K^T / 32, causal-masked, batched over B*H = 32 ----
    {
        dim3 grid(Sc / BN, Sc / BM, Bc * Hc);
        gemm_wmma<0,1><<<grid, blk, 0, stream>>>(qb, kb, attn_out, nullptr, nullptr,
            Sc, Sc, DEPTHc, DEPTHc, DEPTHc,
            (long long)Sc * DEPTHc, (long long)Sc * DEPTHc, (long long)Sc * Sc,
            1.0f / 32.0f);
    }

    // ---- softmax over 65536 rows of 2048 (in place in d_out) ----
    softmax_rows<<<Bc * Hc * Sc, blk, 0, stream>>>(attn_out);

    // ---- ctx = attn @ V ; A f32 converted during staging, B = V^T (N-major) ----
    {
        dim3 grid(1, Sc / BM, Bc * Hc);   // N = 64 handled by bounds checks
        gemm_wmma<1,2><<<grid, blk, 0, stream>>>(attn_out, vt, ctxb, nullptr, nullptr,
            Sc, DEPTHc, Sc, Sc, Sc,
            (long long)Sc * Sc, (long long)DEPTHc * Sc, 0, 0.0f);
    }

    // ---- attn projection: ctx @ Wo + bo + x (residual) -> h1 f32 ----
    {
        dim3 grid(Dc / BN, BSc / BM, 1);
        gemm_wmma<0,3><<<grid, blk, 0, stream>>>(ctxb, Wot, h1, bo, x,
            BSc, Dc, Dc, Dc, Dc, 0, 0, 0, 0.0f);
    }

    // ---- LN1 -> x1 (f32 for residual, bf16 for FFN GEMM) ----
    layernorm_rows<<<BSc, blk, 0, stream>>>(h1, ln1g, ln1b, x1f, x1b);

    // ---- FFN up: gelu(x1 @ W1 + b1) -> gb bf16 ----
    {
        dim3 grid(DFFc / BN, BSc / BM, 1);
        gemm_wmma<0,4><<<grid, blk, 0, stream>>>(x1b, W1t, gb, b1, nullptr,
            BSc, DFFc, Dc, Dc, Dc, 0, 0, 0, 0.0f);
    }

    // ---- FFN down: gb @ W2 + b2 + x1 (residual) -> h2 f32 ----
    {
        dim3 grid(Dc / BN, BSc / BM, 1);
        gemm_wmma<0,3><<<grid, blk, 0, stream>>>(gb, W2t, h2, b2, x1f,
            BSc, Dc, DFFc, DFFc, DFFc, 0, 0, 0, 0.0f);
    }

    // ---- LN2 -> x2 (d_out) ----
    layernorm_rows<<<BSc, blk, 0, stream>>>(h2, ln2g, ln2b, x2_out, nullptr);
}
